// TestModel_31473520345942
// MI455X (gfx1250) — compile-verified, matrix operand order checked by reading, not run on hardware
//
#include <hip/hip_runtime.h>
#include <hip/hip_bf16.h>

typedef __attribute__((ext_vector_type(16))) int   v16i;
typedef __attribute__((ext_vector_type(8)))  float v8f;

#define T_DIM 4096
#define H_DIM 4096
#define BLK   128
#define KBLKS (H_DIM / BLK)   // 32
#define FP8_MAX 448.0f

// ---------------------------------------------------------------------------
// fp32 -> e4m3: use v_cvt_pk_fp8_f32 when available, else manual fallback
// ---------------------------------------------------------------------------
__device__ inline unsigned char cvt_e4m3(float f) {
    f = fminf(fmaxf(f, -FP8_MAX), FP8_MAX);
    unsigned u = __float_as_uint(f);
    unsigned s = (u >> 31) << 7;
    unsigned a = u & 0x7FFFFFFFu;
    if (a < 0x3C000000u) return (unsigned char)s;
    a += 0x000FFFFFu + ((a >> 20) & 1u);
    int e = (int)(a >> 23) - 127 + 7;
    unsigned m = (a >> 20) & 0x7u;
    if (e > 15) { e = 15; m = 7; }
    if (e < 0)  return (unsigned char)s;
    return (unsigned char)(s | ((unsigned)e << 3) | m);
}

__device__ inline unsigned pack4_e4m3(float a, float b, float c, float d) {
    a = fminf(fmaxf(a, -FP8_MAX), FP8_MAX);
    b = fminf(fmaxf(b, -FP8_MAX), FP8_MAX);
    c = fminf(fmaxf(c, -FP8_MAX), FP8_MAX);
    d = fminf(fmaxf(d, -FP8_MAX), FP8_MAX);
#if __has_builtin(__builtin_amdgcn_cvt_pk_fp8_f32)
    int lo = __builtin_amdgcn_cvt_pk_fp8_f32(a, b, 0, false);   // bytes 0..1
    int hi = __builtin_amdgcn_cvt_pk_fp8_f32(c, d, lo, true);   // bytes 2..3
    return (unsigned)hi;
#else
    return (unsigned)cvt_e4m3(a) | ((unsigned)cvt_e4m3(b) << 8) |
           ((unsigned)cvt_e4m3(c) << 16) | ((unsigned)cvt_e4m3(d) << 24);
#endif
}

// ---------------------------------------------------------------------------
// CDNA5 async global -> LDS copy (ASYNCcnt path), 16B per lane per op
// ---------------------------------------------------------------------------
__device__ inline void async_copy16(unsigned lds_off, const unsigned char* gptr) {
    asm volatile("global_load_async_to_lds_b128 %0, %1, off"
                 :
                 : "v"(lds_off), "v"((unsigned long long)gptr)
                 : "memory");
}
__device__ inline void wait_async0() {
    asm volatile("s_wait_asynccnt 0x0" ::: "memory");
}

// ---------------------------------------------------------------------------
// block-wide sum reduction (wave32 shfl + LDS)
// ---------------------------------------------------------------------------
__device__ inline float block_reduce_sum(float v) {
    __shared__ float red[8];
    for (int o = 16; o > 0; o >>= 1) v += __shfl_xor(v, o, 32);
    const int lane = threadIdx.x & 31, w = threadIdx.x >> 5;
    if (lane == 0) red[w] = v;
    __syncthreads();
    if (w == 0) {
        float t = (lane < 8) ? red[lane] : 0.0f;
        for (int o = 4; o > 0; o >>= 1) t += __shfl_xor(t, o, 32);
        if (lane == 0) red[0] = t;
    }
    __syncthreads();
    return red[0];
}

// ---------------------------------------------------------------------------
// 0) one-time: fp32 weights -> e4m3 bytes
// ---------------------------------------------------------------------------
__global__ __launch_bounds__(256)
void k_pack_weights(const float* __restrict__ w, unsigned char* __restrict__ wq, long n4) {
    long i = (long)blockIdx.x * blockDim.x + threadIdx.x;
    if (i >= n4) return;
    const float4 v = ((const float4*)w)[i];
    ((unsigned*)wq)[i] = pack4_e4m3(v.x, v.y, v.z, v.w);
}

// ---------------------------------------------------------------------------
// 1) resid = relu(x); y = rmsnorm(resid, nw)    (one block per row)
// ---------------------------------------------------------------------------
__global__ __launch_bounds__(256)
void k_relu_rmsnorm(const float* __restrict__ x, const float* __restrict__ nw,
                    float* __restrict__ resid, float* __restrict__ y) {
    const int row = blockIdx.x;
    const size_t base = (size_t)row * H_DIM;
    float ss = 0.0f;
    for (int c = threadIdx.x; c < H_DIM; c += 256) {
        float v = fmaxf(x[base + c], 0.0f);
        resid[base + c] = v;
        ss += v * v;
    }
    ss = block_reduce_sum(ss);
    const float r = rsqrtf(ss / (float)H_DIM + 1e-6f);
    for (int c = threadIdx.x; c < H_DIM; c += 256)
        y[base + c] = resid[base + c] * r * nw[c];
}

// ---------------------------------------------------------------------------
// 2) dynamic per-(1,128)-group FP8 quant; scales transposed: ascale_t[kb*T+row]
// ---------------------------------------------------------------------------
__global__ __launch_bounds__(256)
void k_quant_act(const float* __restrict__ y, unsigned char* __restrict__ aq,
                 float* __restrict__ ascale_t) {
    const int gid  = blockIdx.x * 8 + (threadIdx.x >> 5);
    const int lane = threadIdx.x & 31;
    const int row  = gid / KBLKS;
    const int kb   = gid % KBLKS;
    const size_t off = (size_t)row * H_DIM + (size_t)kb * BLK + lane * 4;
    const float4 v = *(const float4*)(y + off);
    float m = fmaxf(fmaxf(fabsf(v.x), fabsf(v.y)), fmaxf(fabsf(v.z), fabsf(v.w)));
    for (int o = 16; o > 0; o >>= 1) m = fmaxf(m, __shfl_xor(m, o, 32));
    const float scale = fmaxf(m / FP8_MAX, 1e-10f);
    const float inv   = 1.0f / scale;
    *(unsigned*)(aq + off) = pack4_e4m3(v.x * inv, v.y * inv, v.z * inv, v.w * inv);
    if (lane == 0) ascale_t[(size_t)kb * T_DIM + row] = scale;
}

// ---------------------------------------------------------------------------
// 3) FP8 block-scaled GEMM, async double-buffered LDS pipeline.
//    Block tile 128(M) x 128(N); 8 waves, each 32(M) x 64(N) = 8 WMMAs/K-step.
//    K-step 128 = one quant block = one v_wmma_f32_16x16x128_fp8_fp8.
// ---------------------------------------------------------------------------
__global__ __launch_bounds__(256)
void k_gemm_fp8(const unsigned char* __restrict__ aq,     // [T,K] e4m3
                const float* __restrict__ ascale_t,       // [K/128, T]
                const unsigned char* __restrict__ wq,     // [N,K] e4m3
                const float* __restrict__ wscale,         // [N/128, K/128]
                float* __restrict__ out)                  // [T,N]
{
    __shared__ __align__(128) unsigned char lA[2][128 * BLK];  // 2 x 16 KB
    __shared__ __align__(128) unsigned char lB[2][128 * BLK];  // 2 x 16 KB

    const int tid  = threadIdx.x;
    const int lane = tid & 31;
    const int wave = tid >> 5;
    const int m0 = blockIdx.x * 128;
    const int n0 = blockIdx.y * 128;
    const int wm = wave >> 1;            // 0..3 -> 32 rows each
    const int wn = wave & 1;             // 0..1 -> 64 cols each
    const int mrow = m0 + wm * 32;
    const int half = lane >> 4;          // K-half / M-half selector

    // staging geometry: thread t covers row = t>>1, 64B half = t&1
    const int sr = tid >> 1, sh = tid & 1;
    const unsigned char* gA = aq + (size_t)(m0 + sr) * H_DIM + sh * 64;
    const unsigned char* gB = wq + (size_t)(n0 + sr) * H_DIM + sh * 64;
    const int lo = sr * BLK + sh * 64;

    v8f acc[2][4];
    for (int a = 0; a < 2; ++a)
        for (int b = 0; b < 4; ++b)
            acc[a][b] = (v8f){0,0,0,0,0,0,0,0};

    // ---- preload K-block 0 into buffer 0 ----
    {
        unsigned la = (unsigned)(unsigned long long)&lA[0][lo];
        unsigned lb = (unsigned)(unsigned long long)&lB[0][lo];
        async_copy16(la +  0, gA +  0); async_copy16(la + 16, gA + 16);
        async_copy16(la + 32, gA + 32); async_copy16(la + 48, gA + 48);
        async_copy16(lb +  0, gB +  0); async_copy16(lb + 16, gB + 16);
        async_copy16(lb + 32, gB + 32); async_copy16(lb + 48, gB + 48);
    }

    for (int kb = 0; kb < KBLKS; ++kb) {
        const int buf = kb & 1;
        wait_async0();          // this wave's in-flight tile is resident
        __syncthreads();        // all waves' tiles resident; prev reads done

        // ---- issue async stage of NEXT K-block into the other buffer ----
        if (kb + 1 < KBLKS) {
            const int nbuf = buf ^ 1;
            const unsigned char* ga = gA + (size_t)(kb + 1) * BLK;
            const unsigned char* gb = gB + (size_t)(kb + 1) * BLK;
            unsigned la = (unsigned)(unsigned long long)&lA[nbuf][lo];
            unsigned lb = (unsigned)(unsigned long long)&lB[nbuf][lo];
            async_copy16(la +  0, ga +  0); async_copy16(la + 16, ga + 16);
            async_copy16(la + 32, ga + 32); async_copy16(la + 48, ga + 48);
            async_copy16(lb +  0, gb +  0); async_copy16(lb + 16, gb + 16);
            async_copy16(lb + 32, gb + 32); async_copy16(lb + 48, gb + 48);
        }

        const float wsc = wscale[blockIdx.y * KBLKS + kb];

        // ---- B fragments (4 x 16 VGPRs) ----
        v16i bfr[4];
        for (int ns = 0; ns < 4; ++ns)
            bfr[ns] = *(const v16i*)&lB[buf][(wn * 64 + ns * 16 + (lane & 15)) * BLK + half * 64];

        // ---- per M-subtile: A fragment + scales, 4 WMMAs each ----
        for (int ms = 0; ms < 2; ++ms) {
            v16i afr = *(const v16i*)&lA[buf][(wm * 32 + ms * 16 + (lane & 15)) * BLK + half * 64];
            const float4* sp = (const float4*)(ascale_t + (size_t)kb * T_DIM + mrow + ms * 16 + half * 8);
            const float4 s0 = sp[0], s1 = sp[1];
            float sa[8];
            sa[0] = s0.x * wsc; sa[1] = s0.y * wsc; sa[2] = s0.z * wsc; sa[3] = s0.w * wsc;
            sa[4] = s1.x * wsc; sa[5] = s1.y * wsc; sa[6] = s1.z * wsc; sa[7] = s1.w * wsc;
            for (int ns = 0; ns < 4; ++ns) {
                v8f z = (v8f){0,0,0,0,0,0,0,0};
                v8f d = __builtin_amdgcn_wmma_f32_16x16x128_fp8_fp8(
                    afr, bfr[ns], (short)0, z, false, false);
                v8f a = acc[ms][ns];
                a[0] = fmaf(d[0], sa[0], a[0]);
                a[1] = fmaf(d[1], sa[1], a[1]);
                a[2] = fmaf(d[2], sa[2], a[2]);
                a[3] = fmaf(d[3], sa[3], a[3]);
                a[4] = fmaf(d[4], sa[4], a[4]);
                a[5] = fmaf(d[5], sa[5], a[5]);
                a[6] = fmaf(d[6], sa[6], a[6]);
                a[7] = fmaf(d[7], sa[7], a[7]);
                acc[ms][ns] = a;
            }
        }
    }

    // ---- epilogue: C layout (VGPR r = M row within half, lane&15 = N) ----
    for (int ms = 0; ms < 2; ++ms) {
        for (int ns = 0; ns < 4; ++ns) {
            const int col = n0 + wn * 64 + ns * 16 + (lane & 15);
            const int rbase = mrow + ms * 16 + half * 8;
            for (int r = 0; r < 8; ++r)
                out[(size_t)(rbase + r) * H_DIM + col] = acc[ms][ns][r];
        }
    }
}

// ---------------------------------------------------------------------------
// 4) resid += xi; y = rmsnorm(resid, nw)     (one block per row)
// ---------------------------------------------------------------------------
__global__ __launch_bounds__(256)
void k_add_rmsnorm(const float* __restrict__ xi, float* __restrict__ resid,
                   const float* __restrict__ nw, float* __restrict__ y) {
    const int row = blockIdx.x;
    const size_t base = (size_t)row * H_DIM;
    float ss = 0.0f;
    for (int c = threadIdx.x; c < H_DIM; c += 256) {
        float v = xi[base + c] + resid[base + c];
        resid[base + c] = v;
        ss += v * v;
    }
    ss = block_reduce_sum(ss);
    const float r = rsqrtf(ss / (float)H_DIM + 1e-6f);
    for (int c = threadIdx.x; c < H_DIM; c += 256)
        y[base + c] = resid[base + c] * r * nw[c];
}

// ---------------------------------------------------------------------------
extern "C" void kernel_launch(void* const* d_in, const int* in_sizes, int n_in,
                              void* d_out, int out_size, void* d_ws, size_t ws_size,
                              hipStream_t stream) {
    const float* x      = (const float*)d_in[0];   // [4096,4096]
    const float* norm_w = (const float*)d_in[1];   // [4,4096]
    const float* w      = (const float*)d_in[2];   // [3,4096,4096]
    const float* wscale = (const float*)d_in[3];   // [3,32,32]
    float* out = (float*)d_out;

    char* ws = (char*)d_ws;
    unsigned char* wq = (unsigned char*)ws;  ws += (size_t)3 * H_DIM * H_DIM;       // 48 MB
    float* resid    = (float*)ws;            ws += (size_t)T_DIM * H_DIM * 4;       // 64 MB
    float* y        = (float*)ws;            ws += (size_t)T_DIM * H_DIM * 4;       // 64 MB
    float* gout     = (float*)ws;            ws += (size_t)T_DIM * H_DIM * 4;       // 64 MB
    unsigned char* aq = (unsigned char*)ws;  ws += (size_t)T_DIM * H_DIM;           // 16 MB
    float* ascale_t = (float*)ws;            ws += (size_t)KBLKS * T_DIM * 4;       // 512 KB

    const long w4 = (long)3 * H_DIM * H_DIM / 4;
    k_pack_weights<<<(unsigned)((w4 + 255) / 256), 256, 0, stream>>>(w, wq, w4);
    k_relu_rmsnorm<<<T_DIM, 256, 0, stream>>>(x, norm_w, resid, y);

    for (int i = 0; i < 3; ++i) {
        k_quant_act<<<T_DIM * KBLKS / 8, 256, 0, stream>>>(y, aq, ascale_t);
        dim3 grid(T_DIM / 128, H_DIM / 128);
        k_gemm_fp8<<<grid, 256, 0, stream>>>(
            aq, ascale_t,
            wq + (size_t)i * H_DIM * H_DIM,
            wscale + (size_t)i * KBLKS * KBLKS,
            gout);
        float* ydst = (i == 2) ? out : y;
        k_add_rmsnorm<<<T_DIM, 256, 0, stream>>>(gout, resid, norm_w + (size_t)(i + 1) * H_DIM, ydst);
    }
}
